// MessageLayer_85718957294176
// MI455X (gfx1250) — matrix operand order repaired and checked by reading, not compiled, not run on hardware
//
#include <hip/hip_runtime.h>
#include <hip/hip_bf16.h>

typedef __attribute__((ext_vector_type(16))) _Float16 v16h;
typedef __attribute__((ext_vector_type(8)))  _Float16 v8h;
typedef __attribute__((ext_vector_type(8)))  float    v8f;

#define DIM 64
#define HEADS 4
#define HD 256          // HEADS*DIM
#define SLOPE 0.2f
#define LN_EPS 1e-5f

// ---------- small helpers ----------
__device__ __forceinline__ unsigned enc_f32(float f) {
    unsigned u = __float_as_uint(f);
    return (u & 0x80000000u) ? ~u : (u | 0x80000000u);
}
__device__ __forceinline__ float dec_f32(unsigned u) {
    unsigned b = (u & 0x80000000u) ? (u & 0x7FFFFFFFu) : ~u;
    return __uint_as_float(b);
}
__device__ __forceinline__ float lrelu(float z) { return z > 0.f ? z : SLOPE * z; }

// ---------- init / convert ----------
__global__ void zero_f32(float* p, long n) {
    long i = (long)blockIdx.x * blockDim.x + threadIdx.x;
    long stride = (long)gridDim.x * blockDim.x;
    for (; i < n; i += stride) p[i] = 0.0f;
}

__global__ void cvt_f32_f16(const float* __restrict__ in, _Float16* __restrict__ out, long n) {
    long i = (long)blockIdx.x * blockDim.x + threadIdx.x;
    long stride = (long)gridDim.x * blockDim.x;
    for (; i < n; i += stride) out[i] = (_Float16)in[i];
}

// ---------- pack weights into per-lane WMMA B fragments (fused f32->f16) ----------
// b[2v]=B[k,col], b[2v+1]=B[k+1,col], k=kc*32+2v+16*(l>>4), col=nt*16+(l&15);
// each lane's 16 halves stored contiguously -> one 32B vector load per chunk.
__global__ void pack_B(const float* __restrict__ src, _Float16* __restrict__ dst,
                       int Kdim, int Nc) {
    int t = blockIdx.x * blockDim.x + threadIdx.x;
    int nch = Kdim >> 5;
    int total = (Nc >> 4) * nch * 256;     // dwords
    if (t >= total) return;
    int v    = t & 7;
    int lane = (t >> 3) & 31;
    int blk  = t >> 8;                     // nt*nch + kc
    int kc   = blk % nch, nt = blk / nch;
    int col  = (nt << 4) + (lane & 15);
    int k    = (kc << 5) + 2 * v + ((lane >> 4) << 4);
    dst[(size_t)t * 2 + 0] = (_Float16)src[(size_t)k * Nc + col];
    dst[(size_t)t * 2 + 1] = (_Float16)src[(size_t)(k + 1) * Nc + col];
}

// ---------- fragment loaders ----------
__device__ __forceinline__ v16h load_a_frag(const _Float16* __restrict__ Arow,
                                            int kc, int hl) {
    const v8h* Ap = reinterpret_cast<const v8h*>(Arow);
    v8h lo = Ap[kc * 4 + 2 * hl];
    v8h hi = Ap[kc * 4 + 2 * hl + 1];
    v16h a;
#pragma unroll
    for (int i = 0; i < 8; ++i) { a[i] = lo[i]; a[8 + i] = hi[i]; }
    return a;
}
__device__ __forceinline__ v16h load_b_frag(const _Float16* __restrict__ Bp,
                                            int frag, int lane) {
    return *reinterpret_cast<const v16h*>(Bp + (((size_t)frag * 32 + lane) << 4));
}

// ---------- fused projection: xl = x@Wl+bl, xr = x@Wr+br (shared A fragments) ----
template <int K, int Nc>
__global__ void wmma_proj(const _Float16* __restrict__ A,
                          const _Float16* __restrict__ Bpl, const _Float16* __restrict__ Bpr,
                          const float* __restrict__ biasL, const float* __restrict__ biasR,
                          float* __restrict__ outL, float* __restrict__ outR, int M) {
    constexpr int NCH = K / 32;
    constexpr int Nt = Nc / 16;
    const int wid = blockIdx.x * (blockDim.x >> 5) + (threadIdx.x >> 5);
    const int Mt = (M + 15) >> 4;
    if (wid >= Mt * Nt) return;
    const int mt = wid / Nt, nt = wid % Nt;
    const int m0 = mt << 4;
    const int lane = threadIdx.x & 31;
    const int hl = lane >> 4, lm = lane & 15;
    int arow = m0 + lm; if (arow >= M) arow = M - 1;
    const _Float16* Arow = A + (size_t)arow * K;
    const int bcol = (nt << 4) + lm;

    v8f cl = {}, cr = {};
#pragma unroll
    for (int kc = 0; kc < NCH; ++kc) {
        v16h a  = load_a_frag(Arow, kc, hl);
        v16h bl = load_b_frag(Bpl, nt * NCH + kc, lane);
        v16h br = load_b_frag(Bpr, nt * NCH + kc, lane);
        cl = __builtin_amdgcn_wmma_f32_16x16x32_f16(false, a, false, bl, (short)0, cl,
                                                    false, false);
        cr = __builtin_amdgcn_wmma_f32_16x16x32_f16(false, a, false, br, (short)0, cr,
                                                    false, false);
    }
    float bL = biasL[bcol], bR = biasR[bcol];
    float* pL = outL + (size_t)(m0 + (hl << 3)) * Nc + bcol;
    float* pR = outR + (size_t)(m0 + (hl << 3)) * Nc + bcol;
    if (m0 + 16 <= M) {                     // full tile: constant-offset stores
#pragma unroll
        for (int r = 0; r < 8; ++r) {
            pL[r * Nc] = cl[r] + bL;
            pR[r * Nc] = cr[r] + bR;
        }
    } else {                                // ragged edge tile
#pragma unroll
        for (int r = 0; r < 8; ++r) {
            if (m0 + (hl << 3) + r < M) {
                pL[r * Nc] = cl[r] + bL;
                pR[r * Nc] = cr[r] + bR;
            }
        }
    }
}

// ---------- generic WMMA GEMM with packed B ----------
template <int K, int Nc, bool ACT, bool OUTH>
__global__ void wmma_gemm(const _Float16* __restrict__ A, const _Float16* __restrict__ Bp,
                          const float* __restrict__ bias,
                          float* __restrict__ outF, _Float16* __restrict__ outH, int M) {
    constexpr int NCH = K / 32;
    constexpr int Nt = Nc / 16;
    const int wid = blockIdx.x * (blockDim.x >> 5) + (threadIdx.x >> 5);
    const int Mt = (M + 15) >> 4;
    if (wid >= Mt * Nt) return;
    const int mt = wid / Nt, nt = wid % Nt;
    const int m0 = mt << 4;
    const int lane = threadIdx.x & 31;
    const int hl = lane >> 4, lm = lane & 15;
    int arow = m0 + lm; if (arow >= M) arow = M - 1;
    const _Float16* Arow = A + (size_t)arow * K;
    const int bcol = (nt << 4) + lm;

    v8f c = {};
#pragma unroll
    for (int kc = 0; kc < NCH; ++kc) {
        v16h a = load_a_frag(Arow, kc, hl);
        v16h b = load_b_frag(Bp, nt * NCH + kc, lane);
        c = __builtin_amdgcn_wmma_f32_16x16x32_f16(false, a, false, b, (short)0, c,
                                                   false, false);
    }
    float bv = bias[bcol];
    size_t pbase = (size_t)(m0 + (hl << 3)) * Nc + bcol;
    if (m0 + 16 <= M) {
#pragma unroll
        for (int r = 0; r < 8; ++r) {
            float v = c[r] + bv;
            if (ACT) v = lrelu(v);
            if (OUTH) outH[pbase + r * Nc] = (_Float16)v;
            else      outF[pbase + r * Nc] = v;
        }
    } else {
#pragma unroll
        for (int r = 0; r < 8; ++r) {
            if (m0 + (hl << 3) + r < M) {
                float v = c[r] + bv;
                if (ACT) v = lrelu(v);
                if (OUTH) outH[pbase + r * Nc] = (_Float16)v;
                else      outF[pbase + r * Nc] = v;
            }
        }
    }
}

// ---------- self-loop edge_attr mean ----------
__global__ void loop_attr_sum(const int* __restrict__ ei, const float* __restrict__ ea,
                              float* __restrict__ lattr, float* __restrict__ cnt, int E) {
    int e = blockIdx.x * blockDim.x + threadIdx.x;
    if (e >= E) return;
    int dst = ei[E + e];
    atomicAdd(&lattr[dst * 3 + 0], ea[e * 3 + 0]);
    atomicAdd(&lattr[dst * 3 + 1], ea[e * 3 + 1]);
    atomicAdd(&lattr[dst * 3 + 2], ea[e * 3 + 2]);
    atomicAdd(&cnt[dst], 1.0f);
}

__global__ void loop_attr_fin(float* __restrict__ lattr, const float* __restrict__ cnt, int N) {
    int n = blockIdx.x * blockDim.x + threadIdx.x;
    if (n >= N) return;
    float c = fmaxf(cnt[n], 1.0f);
    lattr[n * 3 + 0] /= c;
    lattr[n * 3 + 1] /= c;
    lattr[n * 3 + 2] /= c;
}

// ---------- edge logits + segment max (b128 gathers + prefetch) ----------
__global__ void edge_logits(const int* __restrict__ ei, const float* __restrict__ ea,
                            const float* __restrict__ lattr,
                            const float* __restrict__ xl, const float* __restrict__ xr,
                            const float* __restrict__ We, const float* __restrict__ att_w,
                            float* __restrict__ logits, unsigned* __restrict__ mEnc,
                            int E, int N) {
    int tid = blockIdx.x * blockDim.x + threadIdx.x;
    int total = (E + N) * HEADS;
    if (tid >= total) return;
    int e = tid >> 2, h = tid & 3;
    int src, dst; float a0, a1, a2;
    if (e < E) {
        src = ei[e]; dst = ei[E + e];
        a0 = ea[e * 3 + 0]; a1 = ea[e * 3 + 1]; a2 = ea[e * 3 + 2];
    } else {
        src = dst = e - E;
        a0 = lattr[src * 3 + 0]; a1 = lattr[src * 3 + 1]; a2 = lattr[src * 3 + 2];
    }
    const float4* pl = reinterpret_cast<const float4*>(xl + (size_t)src * HD + h * DIM);
    const float4* pr = reinterpret_cast<const float4*>(xr + (size_t)dst * HD + h * DIM);
    const float4* w0 = reinterpret_cast<const float4*>(We + 0 * HD + h * DIM);
    const float4* w1 = reinterpret_cast<const float4*>(We + 1 * HD + h * DIM);
    const float4* w2 = reinterpret_cast<const float4*>(We + 2 * HD + h * DIM);
    const float4* aw = reinterpret_cast<const float4*>(att_w + h * DIM);
    __builtin_prefetch(pl, 0, 0);          // global_prefetch_b8 on the gather streams
    __builtin_prefetch(pr, 0, 0);
    float logit = 0.f;
#pragma unroll 4
    for (int i = 0; i < DIM / 4; ++i) {
        float4 l = pl[i], r = pr[i], e0 = w0[i], e1 = w1[i], e2 = w2[i], w = aw[i];
        logit += lrelu(l.x + r.x + a0 * e0.x + a1 * e1.x + a2 * e2.x) * w.x;
        logit += lrelu(l.y + r.y + a0 * e0.y + a1 * e1.y + a2 * e2.y) * w.y;
        logit += lrelu(l.z + r.z + a0 * e0.z + a1 * e1.z + a2 * e2.z) * w.z;
        logit += lrelu(l.w + r.w + a0 * e0.w + a1 * e1.w + a2 * e2.w) * w.w;
    }
    logits[(size_t)e * HEADS + h] = logit;
    atomicMax(&mEnc[dst * HEADS + h], enc_f32(logit));
}

// ---------- exp + segment denom ----------
__global__ void edge_expsum(const int* __restrict__ ei, float* __restrict__ logits,
                            const unsigned* __restrict__ mEnc, float* __restrict__ denom,
                            int E, int N) {
    int tid = blockIdx.x * blockDim.x + threadIdx.x;
    int total = (E + N) * HEADS;
    if (tid >= total) return;
    int e = tid >> 2, h = tid & 3;
    int dst = (e < E) ? ei[E + e] : (e - E);
    float m = dec_f32(mEnc[dst * HEADS + h]);
    float ev = __expf(logits[tid] - m);
    logits[tid] = ev;
    atomicAdd(&denom[dst * HEADS + h], ev);
}

// ---------- weighted scatter-add: acc[dst,c] += sum_h alpha*xl[src,h,c] ----------
__global__ void edge_aggregate(const int* __restrict__ ei, const float* __restrict__ evals,
                               const float* __restrict__ denom, const float* __restrict__ xl,
                               float* __restrict__ acc, int E, int N) {
    long tid = (long)blockIdx.x * blockDim.x + threadIdx.x;
    long total = (long)(E + N) * DIM;
    if (tid >= total) return;
    int e = (int)(tid >> 6), c = (int)(tid & 63);
    int src, dst;
    if (e < E) { src = ei[e]; dst = ei[E + e]; } else { src = dst = e - E; }
    float4 ev = *reinterpret_cast<const float4*>(evals + (size_t)e * HEADS);
    float4 dn = *reinterpret_cast<const float4*>(denom + (size_t)dst * HEADS);
    const float* px = xl + (size_t)src * HD + c;
    float s = (ev.x / dn.x) * px[0 * DIM] + (ev.y / dn.y) * px[1 * DIM] +
              (ev.z / dn.z) * px[2 * DIM] + (ev.w / dn.w) * px[3 * DIM];
    atomicAdd(&acc[(size_t)dst * DIM + c], s);
}

// ---------- residual + LayerNorm (wave32: 2 channels/lane, shuffle tree) ----------
__global__ void ln1_kernel(const float* __restrict__ x, const float* __restrict__ acc,
                           const float* __restrict__ gat_b, const float* __restrict__ g,
                           const float* __restrict__ b, float* __restrict__ hln,
                           _Float16* __restrict__ h16, int N) {
    int node = blockIdx.x * (blockDim.x >> 5) + (threadIdx.x >> 5);
    if (node >= N) return;
    int lane = threadIdx.x & 31;
    int c0 = lane * 2;
    size_t base = (size_t)node * DIM;
    float2 xv = *reinterpret_cast<const float2*>(x + base + c0);
    float2 av = *reinterpret_cast<const float2*>(acc + base + c0);
    float2 gb = *reinterpret_cast<const float2*>(gat_b + c0);
    float v0 = xv.x + av.x * (1.0f / HEADS) + gb.x;
    float v1 = xv.y + av.y * (1.0f / HEADS) + gb.y;
    float s = v0 + v1;
#pragma unroll
    for (int off = 16; off > 0; off >>= 1) s += __shfl_xor(s, off, 32);
    float mu = s * (1.0f / DIM);
    float d0 = v0 - mu, d1 = v1 - mu;
    float q = d0 * d0 + d1 * d1;
#pragma unroll
    for (int off = 16; off > 0; off >>= 1) q += __shfl_xor(q, off, 32);
    float r = rsqrtf(q * (1.0f / DIM) + LN_EPS);
    float2 gv = *reinterpret_cast<const float2*>(g + c0);
    float2 bv = *reinterpret_cast<const float2*>(b + c0);
    float y0 = d0 * r * gv.x + bv.x;
    float y1 = d1 * r * gv.y + bv.y;
    *reinterpret_cast<float2*>(hln + base + c0) = make_float2(y0, y1);
    h16[base + c0] = (_Float16)y0; h16[base + c0 + 1] = (_Float16)y1;
}

__global__ void ln2_kernel(const float* __restrict__ hln, const float* __restrict__ ffn,
                           const float* __restrict__ g, const float* __restrict__ b,
                           float* __restrict__ out, int N) {
    int node = blockIdx.x * (blockDim.x >> 5) + (threadIdx.x >> 5);
    if (node >= N) return;
    int lane = threadIdx.x & 31;
    int c0 = lane * 2;
    size_t base = (size_t)node * DIM;
    float2 hv = *reinterpret_cast<const float2*>(hln + base + c0);
    float2 fv = *reinterpret_cast<const float2*>(ffn + base + c0);
    float v0 = hv.x + fv.x, v1 = hv.y + fv.y;
    float s = v0 + v1;
#pragma unroll
    for (int off = 16; off > 0; off >>= 1) s += __shfl_xor(s, off, 32);
    float mu = s * (1.0f / DIM);
    float d0 = v0 - mu, d1 = v1 - mu;
    float q = d0 * d0 + d1 * d1;
#pragma unroll
    for (int off = 16; off > 0; off >>= 1) q += __shfl_xor(q, off, 32);
    float r = rsqrtf(q * (1.0f / DIM) + LN_EPS);
    float2 gv = *reinterpret_cast<const float2*>(g + c0);
    float2 bv = *reinterpret_cast<const float2*>(b + c0);
    *reinterpret_cast<float2*>(out + base + c0) =
        make_float2(d0 * r * gv.x + bv.x, d1 * r * gv.y + bv.y);
}

// ---------- host ----------
static inline int cdiv(long a, long b) { return (int)((a + b - 1) / b); }

extern "C" void kernel_launch(void* const* d_in, const int* in_sizes, int n_in,
                              void* d_out, int out_size, void* d_ws, size_t ws_size,
                              hipStream_t stream) {
    const float* x     = (const float*)d_in[0];
    const int*   ei    = (const int*)  d_in[1];
    const float* ea    = (const float*)d_in[2];
    const float* Wl    = (const float*)d_in[3];
    const float* bl    = (const float*)d_in[4];
    const float* Wr    = (const float*)d_in[5];
    const float* br    = (const float*)d_in[6];
    const float* We    = (const float*)d_in[7];
    const float* att_w = (const float*)d_in[8];
    const float* gat_b = (const float*)d_in[9];
    const float* ln1_g = (const float*)d_in[10];
    const float* ln1_b = (const float*)d_in[11];
    const float* ln2_g = (const float*)d_in[12];
    const float* ln2_b = (const float*)d_in[13];
    const float* W1    = (const float*)d_in[14];
    const float* b1    = (const float*)d_in[15];
    const float* W2    = (const float*)d_in[16];
    const float* b2    = (const float*)d_in[17];
    float* out = (float*)d_out;

    const int N = in_sizes[0] / DIM;
    const int E = in_sizes[1] / 2;

    size_t off = 0;
    char* base = (char*)d_ws;
    auto give = [&](size_t bytes) -> char* {
        char* p = base + off;
        off += (bytes + 255) & ~(size_t)255;
        return p;
    };
    float*     xl     = (float*)    give((size_t)N * HD * 4);
    float*     xr     = (float*)    give((size_t)N * HD * 4);
    _Float16*  x16    = (_Float16*) give((size_t)N * DIM * 2);
    _Float16*  Blp    = (_Float16*) give((size_t)DIM * HD * 2);        // packed Wl
    _Float16*  Brp    = (_Float16*) give((size_t)DIM * HD * 2);        // packed Wr
    _Float16*  W1p    = (_Float16*) give((size_t)DIM * 2 * DIM * 2);   // packed W1
    _Float16*  W2p    = (_Float16*) give((size_t)2 * DIM * DIM * 2);   // packed W2
    // contiguous zero region
    char* zstart = base + off;
    float*     lattr  = (float*)    give((size_t)N * 3 * 4);
    float*     cnt    = (float*)    give((size_t)N * 4);
    unsigned*  mEnc   = (unsigned*) give((size_t)N * HEADS * 4);       // 0u == enc(-inf)
    float*     denom  = (float*)    give((size_t)N * HEADS * 4);
    float*     acc    = (float*)    give((size_t)N * DIM * 4);
    char* zend = base + off;
    float*     logits = (float*)    give((size_t)(E + N) * HEADS * 4);
    float*     hln    = (float*)    give((size_t)N * DIM * 4);
    _Float16*  h16    = (_Float16*) give((size_t)N * DIM * 2);
    _Float16*  t16    = (_Float16*) give((size_t)N * 2 * DIM * 2);
    float*     ffn    = (float*)    give((size_t)N * DIM * 4);
    (void)ws_size; (void)n_in; (void)out_size;

    const int B = 256;

    // 1) zero accumulators (single contiguous sweep)
    long zn = (long)(zend - zstart) / 4;
    zero_f32<<<cdiv(zn, B), B, 0, stream>>>((float*)zstart, zn);

    // 2) operand prep: x -> f16; weights -> packed f16 fragments
    cvt_f32_f16<<<cdiv((long)N * DIM, B), B, 0, stream>>>(x, x16, (long)N * DIM);
    pack_B<<<cdiv((long)(HD / 16) * (DIM / 32) * 256, B), B, 0, stream>>>(Wl, Blp, DIM, HD);
    pack_B<<<cdiv((long)(HD / 16) * (DIM / 32) * 256, B), B, 0, stream>>>(Wr, Brp, DIM, HD);
    pack_B<<<cdiv((long)(2 * DIM / 16) * (DIM / 32) * 256, B), B, 0, stream>>>(W1, W1p, DIM, 2 * DIM);
    pack_B<<<cdiv((long)(DIM / 16) * (2 * DIM / 32) * 256, B), B, 0, stream>>>(W2, W2p, 2 * DIM, DIM);

    // 3) self-loop edge_attr = per-target mean
    loop_attr_sum<<<cdiv(E, B), B, 0, stream>>>(ei, ea, lattr, cnt, E);
    loop_attr_fin<<<cdiv(N, B), B, 0, stream>>>(lattr, cnt, N);

    // 4) fused projections via WMMA (shared A fragments): [N,64]@[64,256] x2
    {
        int tiles = ((N + 15) / 16) * (HD / 16);
        wmma_proj<DIM, HD><<<cdiv(tiles, B / 32), B, 0, stream>>>(
            x16, Blp, Brp, bl, br, xl, xr, N);
    }

    // 5-7) edge phase
    edge_logits<<<cdiv((long)(E + N) * HEADS, B), B, 0, stream>>>(
        ei, ea, lattr, xl, xr, We, att_w, logits, mEnc, E, N);
    edge_expsum<<<cdiv((long)(E + N) * HEADS, B), B, 0, stream>>>(
        ei, logits, mEnc, denom, E, N);
    edge_aggregate<<<cdiv((long)(E + N) * DIM, B), B, 0, stream>>>(
        ei, logits, denom, xl, acc, E, N);

    // 8) residual + LN1
    ln1_kernel<<<cdiv(N, B / 32), B, 0, stream>>>(x, acc, gat_b, ln1_g, ln1_b, hln, h16, N);

    // 9) FFN via WMMA
    {
        int tiles1 = ((N + 15) / 16) * (2 * DIM / 16);
        wmma_gemm<DIM, 2 * DIM, true, true><<<cdiv(tiles1, B / 32), B, 0, stream>>>(
            h16, W1p, b1, nullptr, t16, N);
        int tiles2 = ((N + 15) / 16) * (DIM / 16);
        wmma_gemm<2 * DIM, DIM, false, false><<<cdiv(tiles2, B / 32), B, 0, stream>>>(
            t16, W2p, b2, ffn, nullptr, N);
    }

    // 10) residual + LN2 -> output
    ln2_kernel<<<cdiv(N, B / 32), B, 0, stream>>>(hln, ffn, ln2_g, ln2_b, out, N);
}
